// ContractGNN_3539053052010
// MI455X (gfx1250) — matrix operand order repaired and checked by reading, not compiled, not run on hardware
//
#include <hip/hip_runtime.h>
#include <hip/hip_bf16.h>
#include <math.h>

#define N_NODES   100000
#define N_GRAPHS  512
#define HDIM      64
#define NEG_SLOPE 0.2f

typedef __attribute__((ext_vector_type(16))) _Float16 v16h;
typedef __attribute__((ext_vector_type(8)))  float    v8f;

__device__ __forceinline__ float lrelu(float x) { return x > 0.f ? x : NEG_SLOPE * x; }

// float atomic max via integer punning (works for mixed signs when all updates use it)
__device__ __forceinline__ void atomicMaxFloat(float* addr, float val) {
  if (val >= 0.f) atomicMax((int*)addr, __float_as_int(val));
  else            atomicMin((unsigned int*)addr, __float_as_uint(val));
}

// native global_atomic_add_f32 (no CAS loop)
__device__ __forceinline__ void atomicAddF32(float* addr, float val) {
  unsafeAtomicAdd(addr, val);
}

// ---------------------------------------------------------------------------
// GEMM: H[N x 64] = X[N x FIN] @ W[FIN x 64], f32 in, f16 WMMA, f32 accum.
// One wave per 16-row tile; 4 column tiles of 16 (HDIM=64); K loop step 32.
// ---------------------------------------------------------------------------
template <int FIN>
__global__ __launch_bounds__(128)
void gat_gemm_wmma(const float* __restrict__ X, const float* __restrict__ W,
                   float* __restrict__ Hout, int nRowTiles) {
  __shared__ float sW[FIN * HDIM];
  for (int i = threadIdx.x; i < FIN * HDIM; i += blockDim.x) sW[i] = W[i];
  __syncthreads();

  const int wave = threadIdx.x >> 5;
  const int lane = threadIdx.x & 31;
  const int tile = blockIdx.x * 4 + wave;
  if (tile >= nRowTiles) return;

  const int rowBase = tile * 16;
  const int mloc = lane & 15;
  const int half = lane >> 4;

  v8f acc[4];
  #pragma unroll
  for (int t = 0; t < 4; ++t)
    #pragma unroll
    for (int r = 0; r < 8; ++r) acc[t][r] = 0.f;

  for (int kk = 0; kk < FIN; kk += 32) {
    // A fragment: 16x32 f16. Lane half 0: K=kk+[0..7],[16..23]; half 1: +8.
    v16h a;
    const float* xr = X + (size_t)(rowBase + mloc) * FIN + kk + half * 8;
    #pragma unroll
    for (int i = 0; i < 8; ++i) a[i]     = (_Float16)xr[i];
    #pragma unroll
    for (int i = 0; i < 8; ++i) a[8 + i] = (_Float16)xr[16 + i];

    #pragma unroll
    for (int t = 0; t < 4; ++t) {
      // B fragment: 32x16 f16. Lane half 0: K=kk+[0..7],[16..23]; half 1: +8.
      v16h b;
      const int col = t * 16 + mloc;
      #pragma unroll
      for (int i = 0; i < 16; ++i)
        b[i] = (_Float16)sW[(kk + half * 8 + (i & 7) + (i >> 3) * 16) * HDIM + col];
      acc[t] = __builtin_amdgcn_wmma_f32_16x16x32_f16(
          /*neg_a=*/false, a, /*neg_b=*/false, b,
          /*c_mod=*/(short)0, acc[t], /*reuse_a=*/false, /*reuse_b=*/false);
    }
  }

  // C/D layout: VGPR r -> M = r + 8*half, N = lane&15
  #pragma unroll
  for (int t = 0; t < 4; ++t)
    #pragma unroll
    for (int r = 0; r < 8; ++r)
      Hout[(size_t)(rowBase + half * 8 + r) * HDIM + t * 16 + mloc] = acc[t][r];
}

// ---------------------------------------------------------------------------
// e_s[i] = h[i].a_src ; e_d[i] = h[i].a_dst  (one wave per node, shfl reduce)
// ---------------------------------------------------------------------------
__global__ __launch_bounds__(256)
void gat_scores(const float* __restrict__ H, const float* __restrict__ as_,
                const float* __restrict__ ad_, float* __restrict__ es,
                float* __restrict__ ed, int N) {
  const int node = (int)((blockIdx.x * (long long)blockDim.x + threadIdx.x) >> 5);
  const int lane = threadIdx.x & 31;
  if (node >= N) return;
  const float* hr = H + (size_t)node * HDIM;
  float s = hr[lane] * as_[lane] + hr[lane + 32] * as_[lane + 32];
  float d = hr[lane] * ad_[lane] + hr[lane + 32] * ad_[lane + 32];
  #pragma unroll
  for (int off = 16; off > 0; off >>= 1) {
    s += __shfl_xor(s, off, 32);
    d += __shfl_xor(d, off, 32);
  }
  if (lane == 0) { es[node] = s; ed[node] = d; }
}

// ---------------------------------------------------------------------------
// Per-node init: m seeded with self-loop logit, den=0, acc zeroed.
// ---------------------------------------------------------------------------
__global__ __launch_bounds__(256)
void gat_node_init(const float* __restrict__ es, const float* __restrict__ ed,
                   float* __restrict__ m, float* __restrict__ den,
                   float* __restrict__ acc, int N) {
  const int node = (int)((blockIdx.x * (long long)blockDim.x + threadIdx.x) >> 5);
  const int lane = threadIdx.x & 31;
  if (node >= N) return;
  acc[(size_t)node * HDIM + lane]      = 0.f;
  acc[(size_t)node * HDIM + lane + 32] = 0.f;
  if (lane == 0) {
    m[node]   = lrelu(es[node] + ed[node]);  // self-loop logit
    den[node] = 0.f;
  }
}

// ---------------------------------------------------------------------------
// Segment max over edges (one thread per edge; NT loads for streamed ei)
// ---------------------------------------------------------------------------
__global__ __launch_bounds__(256)
void gat_edge_max(const long long* __restrict__ ei, long long E,
                  const float* __restrict__ es, const float* __restrict__ ed,
                  float* __restrict__ m) {
  const long long e = blockIdx.x * (long long)blockDim.x + threadIdx.x;
  if (e >= E) return;
  const int s = (int)__builtin_nontemporal_load(ei + e);
  const int d = (int)__builtin_nontemporal_load(ei + E + e);
  atomicMaxFloat(&m[d], lrelu(es[s] + ed[d]));
}

// ---------------------------------------------------------------------------
// Unnormalized softmax accumulate, blocked: each wave owns 32 edges.
// Lanes load 32 src/dst coalesced + compute ex; then a broadcast loop walks
// the 32 edges with all lanes doing the 64-channel gather/scatter (2 ch/lane).
// acc[dst] += ex * h[src] ; den[dst] += ex
// ---------------------------------------------------------------------------
__global__ __launch_bounds__(256)
void gat_edge_accum(const long long* __restrict__ ei, long long E,
                    const float* __restrict__ es, const float* __restrict__ ed,
                    const float* __restrict__ m, const float* __restrict__ H,
                    float* __restrict__ den, float* __restrict__ acc) {
  const long long waveId = (blockIdx.x * (long long)blockDim.x + threadIdx.x) >> 5;
  const int lane = threadIdx.x & 31;
  const long long base = waveId * 32;
  if (base >= E) return;
  const int cnt = (int)((E - base < 32) ? (E - base) : 32);

  int s = 0, d = 0;
  float ex = 0.f;
  if (lane < cnt) {
    s = (int)__builtin_nontemporal_load(ei + base + lane);
    d = (int)__builtin_nontemporal_load(ei + E + base + lane);
    // get the gather rows in flight before the broadcast loop consumes them
    __builtin_prefetch(H + (size_t)s * HDIM, 0, 3);
    __builtin_prefetch(H + (size_t)s * HDIM + 32, 0, 3);
    ex = __expf(lrelu(es[s] + ed[d]) - m[d]);
    atomicAddF32(&den[d], ex);
  }

  for (int i = 0; i < cnt; ++i) {
    const int   si  = __shfl(s, i, 32);
    const int   di  = __shfl(d, i, 32);
    const float exi = __shfl(ex, i, 32);
    const float* hs = H + (size_t)si * HDIM;
    float* ad2 = acc + (size_t)di * HDIM;
    atomicAddF32(&ad2[lane],      exi * hs[lane]);
    atomicAddF32(&ad2[lane + 32], exi * hs[lane + 32]);
  }
}

// ---------------------------------------------------------------------------
// Epilogue: fold self-loop term, normalize, add bias, optional ReLU.
// ---------------------------------------------------------------------------
__global__ __launch_bounds__(256)
void gat_node_finish(const float* __restrict__ es, const float* __restrict__ ed,
                     const float* __restrict__ m, const float* __restrict__ den,
                     const float* __restrict__ H, const float* __restrict__ acc,
                     const float* __restrict__ bias, float* __restrict__ outF,
                     int N, int doRelu) {
  const int node = (int)((blockIdx.x * (long long)blockDim.x + threadIdx.x) >> 5);
  const int lane = threadIdx.x & 31;
  if (node >= N) return;
  const float exs = __expf(lrelu(es[node] + ed[node]) - m[node]);
  const float tot = fmaxf(den[node] + exs, 1e-16f);
  const float inv = 1.f / tot;
  #pragma unroll
  for (int j = 0; j < 2; ++j) {
    const int c = lane + j * 32;
    float v = (acc[(size_t)node * HDIM + c] + exs * H[(size_t)node * HDIM + c]) * inv + bias[c];
    if (doRelu) v = fmaxf(v, 0.f);
    outF[(size_t)node * HDIM + c] = v;
  }
}

// ---------------------------------------------------------------------------
// Pooling
// ---------------------------------------------------------------------------
__global__ __launch_bounds__(256)
void pool_init(float* __restrict__ sums, float* __restrict__ gmax,
               float* __restrict__ counts, int total) {
  const int i = blockIdx.x * blockDim.x + threadIdx.x;
  if (i < total) { sums[i] = 0.f; gmax[i] = -INFINITY; }
  if (i < N_GRAPHS) counts[i] = 0.f;
}

__global__ __launch_bounds__(256)
void pool_scatter(const float* __restrict__ H, const long long* __restrict__ batch,
                  float* __restrict__ sums, float* __restrict__ gmax,
                  float* __restrict__ counts, int N) {
  const int node = (int)((blockIdx.x * (long long)blockDim.x + threadIdx.x) >> 5);
  const int lane = threadIdx.x & 31;
  if (node >= N) return;
  const long long g = batch[node];
  const float* hr = H + (size_t)node * HDIM;
  #pragma unroll
  for (int j = 0; j < 2; ++j) {
    const int c = lane + j * 32;
    atomicAddF32(&sums[(size_t)g * HDIM + c], hr[c]);
    atomicMaxFloat(&gmax[(size_t)g * HDIM + c], hr[c]);
  }
  if (lane == 0) atomicAddF32(&counts[g], 1.f);
}

__global__ __launch_bounds__(256)
void final_linear(const float* __restrict__ sums, const float* __restrict__ gmax,
                  const float* __restrict__ counts, const float* __restrict__ linW,
                  const float* __restrict__ linb, float* __restrict__ out) {
  const int tid = blockIdx.x * blockDim.x + threadIdx.x;
  if (tid >= N_GRAPHS * 10) return;
  const int g = tid / 10;
  const int k = tid % 10;
  const float cinv = 1.f / fmaxf(counts[g], 1.f);
  float acc = linb[k];
  for (int c = 0; c < HDIM; ++c) {
    const float mean = sums[(size_t)g * HDIM + c] * cinv;
    float mx = gmax[(size_t)g * HDIM + c];
    if (!isfinite(mx)) mx = 0.f;
    acc += mean * linW[c * 10 + k];
    acc += mx * linW[(HDIM + c) * 10 + k];
  }
  out[g * 10 + k] = acc;
}

// ---------------------------------------------------------------------------
// Host launcher
// ---------------------------------------------------------------------------
extern "C" void kernel_launch(void* const* d_in, const int* in_sizes, int n_in,
                              void* d_out, int out_size, void* d_ws, size_t ws_size,
                              hipStream_t stream) {
  const float*     x     = (const float*)d_in[0];
  const long long* ei    = (const long long*)d_in[1];
  const long long* batch = (const long long*)d_in[2];
  const float* W1 = (const float*)d_in[3];
  const float* as1 = (const float*)d_in[4];
  const float* ad1 = (const float*)d_in[5];
  const float* b1 = (const float*)d_in[6];
  const float* W2 = (const float*)d_in[7];
  const float* as2 = (const float*)d_in[8];
  const float* ad2 = (const float*)d_in[9];
  const float* b2 = (const float*)d_in[10];
  const float* W3 = (const float*)d_in[11];
  const float* as3 = (const float*)d_in[12];
  const float* ad3 = (const float*)d_in[13];
  const float* b3 = (const float*)d_in[14];
  const float* linW = (const float*)d_in[15];
  const float* linb = (const float*)d_in[16];

  const int N = N_NODES;
  const long long E = in_sizes[1] / 2;

  // Workspace carve-up
  float* p = (float*)d_ws;
  float* hbuf = p;              p += (size_t)N * HDIM;   // GEMM output h
  float* acc0 = p;              p += (size_t)N * HDIM;   // layer accumulator A
  float* acc1 = p;              p += (size_t)N * HDIM;   // layer accumulator B
  float* es   = p;              p += N;
  float* ed   = p;              p += N;
  float* mbuf = p;              p += N;
  float* den  = p;              p += N;
  float* sums = p;              p += (size_t)N_GRAPHS * HDIM;
  float* gmax = p;              p += (size_t)N_GRAPHS * HDIM;
  float* cnts = p;              p += N_GRAPHS;

  const int nRowTiles  = N / 16;                         // 6250
  const int gemmBlocks = (nRowTiles + 3) / 4;
  const int nodeWaveBlocks = (int)(((long long)N * 32 + 255) / 256);
  const int edgeThreadBlocks = (int)((E + 255) / 256);
  // blocked accum: one wave per 32 edges -> 256 edges per 256-thread block
  const int edgeBlockBlocks = (int)((E + 255) / 256);

  struct Layer {
    const float* feat; const float* W; const float* as_; const float* ad_;
    const float* bias; float* accum; float* outF; int relu;
  };
  Layer L[3] = {
    { x,    W1, as1, ad1, b1, acc0, acc0, 1 },
    { acc0, W2, as2, ad2, b2, acc1, acc1, 1 },
    { acc1, W3, as3, ad3, b3, acc0, acc0, 0 },
  };

  for (int li = 0; li < 3; ++li) {
    if (li == 0)
      gat_gemm_wmma<128><<<gemmBlocks, 128, 0, stream>>>(L[li].feat, L[li].W, hbuf, nRowTiles);
    else
      gat_gemm_wmma<64><<<gemmBlocks, 128, 0, stream>>>(L[li].feat, L[li].W, hbuf, nRowTiles);

    gat_scores<<<nodeWaveBlocks, 256, 0, stream>>>(hbuf, L[li].as_, L[li].ad_, es, ed, N);
    gat_node_init<<<nodeWaveBlocks, 256, 0, stream>>>(es, ed, mbuf, den, L[li].accum, N);
    gat_edge_max<<<edgeThreadBlocks, 256, 0, stream>>>(ei, E, es, ed, mbuf);
    gat_edge_accum<<<edgeBlockBlocks, 256, 0, stream>>>(ei, E, es, ed, mbuf, hbuf, den, L[li].accum);
    gat_node_finish<<<nodeWaveBlocks, 256, 0, stream>>>(es, ed, mbuf, den, hbuf, L[li].accum,
                                                        L[li].bias, L[li].outF, N, L[li].relu);
  }

  pool_init<<<(N_GRAPHS * HDIM + 255) / 256, 256, 0, stream>>>(sums, gmax, cnts, N_GRAPHS * HDIM);
  pool_scatter<<<nodeWaveBlocks, 256, 0, stream>>>(acc0, batch, sums, gmax, cnts, N);
  final_linear<<<(N_GRAPHS * 10 + 255) / 256, 256, 0, stream>>>(sums, gmax, cnts, linW, linb,
                                                                (float*)d_out);
}
